// StateRefiner_13520557048073
// MI455X (gfx1250) — matrix operand order repaired and checked by reading, not compiled, not run on hardware
//
#include <hip/hip_runtime.h>
#include <cstdint>
#include <cstddef>

typedef __bf16 bf16;
typedef __bf16  v16bf __attribute__((ext_vector_type(16)));
typedef float   v8f   __attribute__((ext_vector_type(8)));

#define HEADS 4
#define DIMH  128
#define NPIX  2304
#define BATCH 2
#define QKSCALE 0.08838834764831845f  /* 128^-0.5 */

static __device__ __forceinline__ bf16 f2bf(float f) { return (bf16)f; }

/* ---------------- tiny fold / convert kernels (negligible cost) ------------ */

__global__ void k_matmul(float* C, const float* A, const float* B,
                         int M, int K, int N, float alpha) {
  int idx = blockIdx.x * blockDim.x + threadIdx.x;
  int total = M * N;
  for (; idx < total; idx += gridDim.x * blockDim.x) {
    int m = idx / N, n = idx % N;
    float acc = 0.f;
    for (int k = 0; k < K; ++k) acc += A[m * K + k] * B[k * N + n];
    C[idx] = alpha * acc;
  }
}

__global__ void k_matvec(float* y, const float* A, const float* x,
                         const float* c, int M, int K, float alpha) {
  int m = blockIdx.x * blockDim.x + threadIdx.x;
  if (m >= M) return;
  float acc = 0.f;
  for (int k = 0; k < K; ++k) acc += A[m * K + k] * x[k];
  y[m] = alpha * acc + (c ? c[m] : 0.f);
}

__global__ void k_f32_to_bf16(bf16* dst, const float* src, int n) {
  int i = blockIdx.x * blockDim.x + threadIdx.x;
  for (; i < n; i += gridDim.x * blockDim.x) dst[i] = f2bf(src[i]);
}

/* V_flow = Wvf(512x2) @ prev_flow + bvf, straight per-pixel FMA */
__global__ void k_vflow(bf16* v, const float* flow, const float* Wvf, const float* bvf) {
  int i = blockIdx.x * blockDim.x + threadIdx.x;
  int total = BATCH * 512 * NPIX;
  for (; i < total; i += gridDim.x * blockDim.x) {
    int n = i % NPIX;
    int c = (i / NPIX) % 512;
    int b = i / (512 * NPIX);
    const float* f = flow + (size_t)b * 2 * NPIX;
    v[i] = f2bf(Wvf[c * 2 + 0] * f[n] + Wvf[c * 2 + 1] * f[NPIX + n] + bvf[c]);
  }
}

/* ---------------- generic bf16 WMMA GEMM -----------------------------------
 * C(MxN) = A(MxK,row-major,bf16) @ B(KxN,row-major,bf16) [+bias[m]]
 * epilogue: resid!=null -> outF = resid + gamma*(acc+bias)  (f32)
 *           else        -> outH = bf16(acc+bias)
 * Block tile 128x64, 8 waves (4x2), 32x32 per wave, k-step 32.
 */
#define GBM 128
#define GBN 64

__global__ __launch_bounds__(256)
void k_wmma_gemm(const bf16* __restrict__ A, const bf16* __restrict__ Bm,
                 const float* __restrict__ bias,
                 const float* __restrict__ resid, const float* __restrict__ gamma_ptr,
                 float* __restrict__ outF, bf16* __restrict__ outH,
                 int M, int K, int N, int storeM)
{
  __shared__ alignas(32) bf16 Al[GBM][40]; /* [m][k], k-contiguous */
  __shared__ alignas(32) bf16 Bl[GBN][48]; /* [n][k], transposed so frags are contiguous */

  const int tid  = threadIdx.x;
  const int lane = tid & 31;
  const int wave = tid >> 5;
  const int wm   = wave >> 1;   /* 0..3 */
  const int wn   = wave & 1;    /* 0..1 */
  const int l15  = lane & 15;
  const int hi   = lane >> 4;   /* half-wave id */
  const int mblk = blockIdx.y * GBM;
  const int nblk = blockIdx.x * GBN;

  v8f acc[2][2] = {};

  for (int k0 = 0; k0 < K; k0 += 32) {
    __syncthreads();
    /* stage A: 128 rows x 32 halfs, 16B chunks, zero-pad rows >= M */
    for (int t = tid; t < 512; t += 256) {
      int r = t >> 2, ch = (t & 3) * 8;
      int gm = mblk + r;
      uint4 val = make_uint4(0u, 0u, 0u, 0u);
      if (gm < M) val = *(const uint4*)(A + (size_t)gm * K + k0 + ch);
      *(uint4*)(&Al[r][ch]) = val;
    }
    /* stage B transposed: 32 k-rows x 64 n */
    {
      int kk = tid >> 3, nc = (tid & 7) * 8;
      uint4 val = *(const uint4*)(Bm + (size_t)(k0 + kk) * N + nblk + nc);
      const bf16* h = (const bf16*)&val;
      #pragma unroll
      for (int j = 0; j < 8; ++j) Bl[nc + j][kk] = h[j];
    }
    __syncthreads();

    v16bf af[2], bfm[2];
    #pragma unroll
    for (int mt = 0; mt < 2; ++mt) {           /* A frag 16x32, contiguous k = 16*hi + s */
      int r = wm * 32 + mt * 16 + l15;
      int kb = hi ? 16 : 0;
      union { v16bf v; uint4 q[2]; } u;
      u.q[0] = *(const uint4*)(&Al[r][kb]);
      u.q[1] = *(const uint4*)(&Al[r][kb + 8]);
      af[mt] = u.v;
    }
    #pragma unroll
    for (int nt = 0; nt < 2; ++nt) {           /* B frag 32x16, contiguous k = 16*hi + s */
      int c = wn * 32 + nt * 16 + l15;
      int kb = hi ? 16 : 0;
      union { v16bf v; uint4 q[2]; } u;
      u.q[0] = *(const uint4*)(&Bl[c][kb]);
      u.q[1] = *(const uint4*)(&Bl[c][kb + 8]);
      bfm[nt] = u.v;
    }
    #pragma unroll
    for (int mt = 0; mt < 2; ++mt)
      #pragma unroll
      for (int nt = 0; nt < 2; ++nt)
        acc[mt][nt] = __builtin_amdgcn_wmma_f32_16x16x32_bf16(
            false, af[mt], false, bfm[nt], (short)0, acc[mt][nt], false, false);
  }

  float g = gamma_ptr ? gamma_ptr[0] : 1.0f;
  #pragma unroll
  for (int mt = 0; mt < 2; ++mt) {
    #pragma unroll
    for (int nt = 0; nt < 2; ++nt) {
      int n = nblk + wn * 32 + nt * 16 + l15;
      int mbase = mblk + wm * 32 + mt * 16 + hi * 8;
      #pragma unroll
      for (int i = 0; i < 8; ++i) {
        int m = mbase + i;
        if (m >= storeM) continue;
        float t = acc[mt][nt][i] + (bias ? bias[m] : 0.f);
        size_t o = (size_t)m * N + n;
        if (resid) outF[o] = resid[o] + g * t;
        else       outH[o] = f2bf(t);
      }
    }
  }
}

/* ---------------- fused flash attention, 3 V sources -----------------------
 * grid: (NPIX/64, HEADS, BATCH), 128 threads (4 waves), 16 q-rows per wave.
 * Q,K,V are (512, NPIX) bf16 channel-major per batch; head h = rows h*128..+127.
 */
__global__ __launch_bounds__(128)
void k_flash(const bf16* __restrict__ Q, const bf16* __restrict__ Km,
             const bf16* __restrict__ Vf, const bf16* __restrict__ Vn,
             const bf16* __restrict__ Vi,
             bf16* __restrict__ Of, bf16* __restrict__ On, bf16* __restrict__ Oi)
{
  __shared__ alignas(32) bf16 Kl[32][144];     /* [px][c] transposed */
  __shared__ alignas(32) bf16 Vl[3][128][48];  /* [src][d][px] natural */
  __shared__ alignas(32) bf16 Pl[4][16][32];   /* per-wave P bounce */

  const int tid  = threadIdx.x;
  const int lane = tid & 31;
  const int wave = tid >> 5;
  const int l15  = lane & 15;
  const int hi   = lane >> 4;

  const int b = blockIdx.z, h = blockIdx.y;
  const size_t headOff = ((size_t)b * 512 + (size_t)h * DIMH) * NPIX;
  const bf16* Qh = Q + headOff;
  const bf16* Kh = Km + headOff;
  const bf16* Vh[3] = { Vf + headOff, Vn + headOff, Vi + headOff };
  bf16* Oh[3] = { Of + headOff, On + headOff, Oi + headOff };

  const int q0 = blockIdx.x * 64 + wave * 16;

  /* Q fragments: 4 chunks of 16x32 (channel-K), strided gather (once) */
  v16bf qf[4];
  #pragma unroll
  for (int cc = 0; cc < 4; ++cc) {
    union { v16bf v; bf16 hlf[16]; } u;
    int kb = cc * 32 + (hi ? 16 : 0);
    #pragma unroll
    for (int j = 0; j < 8; ++j) {
      u.hlf[j]     = Qh[(size_t)(kb + j)      * NPIX + q0 + l15];
      u.hlf[j + 8] = Qh[(size_t)(kb + 8 + j) * NPIX + q0 + l15];
    }
    qf[cc] = u.v;
  }

  v8f acc[3][8] = {};
  float rowm[8], rows[8];
  #pragma unroll
  for (int i = 0; i < 8; ++i) { rowm[i] = -1e30f; rows[i] = 0.f; }

  #pragma unroll 1
  for (int kp = 0; kp < NPIX; kp += 32) {
    __syncthreads();
    /* stage K tile transposed (contiguous B-frag reads for QK^T) */
    for (int t = tid; t < 512; t += 128) {
      int c = t >> 2, pxc = (t & 3) * 8;
      uint4 val = *(const uint4*)(Kh + (size_t)c * NPIX + kp + pxc);
      const bf16* hp = (const bf16*)&val;
      #pragma unroll
      for (int j = 0; j < 8; ++j) Kl[pxc + j][c] = hp[j];
    }
    /* stage 3 V tiles naturally (contiguous B-frag reads for P@V) */
    #pragma unroll
    for (int s = 0; s < 3; ++s)
      for (int t = tid; t < 512; t += 128) {
        int d = t >> 2, pxc = (t & 3) * 8;
        *(uint4*)(&Vl[s][d][pxc]) =
            *(const uint4*)(Vh[s] + (size_t)d * NPIX + kp + pxc);
      }
    __syncthreads();

    /* S = Q K^T : two 16x16 px-tiles, K-dim = 128 channels (4 wmma each) */
    v8f s0 = {}, s1 = {};
    #pragma unroll
    for (int cc = 0; cc < 4; ++cc) {
      int cb = cc * 32 + (hi ? 16 : 0);
      union { v16bf v; uint4 q[2]; } u0, u1;
      u0.q[0] = *(const uint4*)(&Kl[l15][cb]);
      u0.q[1] = *(const uint4*)(&Kl[l15][cb + 8]);
      u1.q[0] = *(const uint4*)(&Kl[16 + l15][cb]);
      u1.q[1] = *(const uint4*)(&Kl[16 + l15][cb + 8]);
      s0 = __builtin_amdgcn_wmma_f32_16x16x32_bf16(false, qf[cc], false, u0.v, (short)0, s0, false, false);
      s1 = __builtin_amdgcn_wmma_f32_16x16x32_bf16(false, qf[cc], false, u1.v, (short)0, s1, false, false);
    }

    /* online softmax over the 32 new columns; half-wave xor reductions */
    float sc[8];
    #pragma unroll
    for (int i = 0; i < 8; ++i) {
      float tmax = fmaxf(s0[i], s1[i]);
      #pragma unroll
      for (int off = 8; off >= 1; off >>= 1)
        tmax = fmaxf(tmax, __shfl_xor(tmax, off, 16));
      float nm = fmaxf(rowm[i], tmax);
      float scale = __expf(rowm[i] - nm);
      rowm[i] = nm;
      float p0 = __expf(s0[i] - nm);
      float p1 = __expf(s1[i] - nm);
      s0[i] = p0; s1[i] = p1;
      float ps = p0 + p1;
      #pragma unroll
      for (int off = 8; off >= 1; off >>= 1)
        ps += __shfl_xor(ps, off, 16);
      rows[i] = rows[i] * scale + ps;
      sc[i] = scale;
    }
    #pragma unroll
    for (int s = 0; s < 3; ++s)
      #pragma unroll
      for (int dt = 0; dt < 8; ++dt)
        #pragma unroll
        for (int i = 0; i < 8; ++i)
          acc[s][dt][i] *= sc[i];

    /* P: C-layout -> A-layout via per-wave LDS bounce */
    #pragma unroll
    for (int i = 0; i < 8; ++i) {
      int m = hi * 8 + i;
      Pl[wave][m][l15]      = f2bf(s0[i]);
      Pl[wave][m][16 + l15] = f2bf(s1[i]);
    }
    asm volatile("s_wait_dscnt 0" ::: "memory");
    v16bf pf;
    {
      union { v16bf v; uint4 q[2]; } u;
      int kb = hi ? 16 : 0;
      u.q[0] = *(const uint4*)(&Pl[wave][l15][kb]);
      u.q[1] = *(const uint4*)(&Pl[wave][l15][kb + 8]);
      pf = u.v;
    }
    /* O += P @ V for 3 sources x 8 d-tiles (24 wmma) */
    #pragma unroll
    for (int s = 0; s < 3; ++s)
      #pragma unroll
      for (int dt = 0; dt < 8; ++dt) {
        union { v16bf v; uint4 q[2]; } u;
        int d = dt * 16 + l15;
        int pb = hi ? 16 : 0;
        u.q[0] = *(const uint4*)(&Vl[s][d][pb]);
        u.q[1] = *(const uint4*)(&Vl[s][d][pb + 8]);
        acc[s][dt] = __builtin_amdgcn_wmma_f32_16x16x32_bf16(
            false, pf, false, u.v, (short)0, acc[s][dt], false, false);
      }
  }

  /* normalize + store O (channel-major bf16), vectorized 16B stores */
  float rinv[8];
  #pragma unroll
  for (int i = 0; i < 8; ++i) rinv[i] = rows[i] > 0.f ? 1.f / rows[i] : 0.f;
  #pragma unroll
  for (int s = 0; s < 3; ++s)
    #pragma unroll
    for (int dt = 0; dt < 8; ++dt) {
      union { uint4 q; bf16 hlf[8]; } u;
      #pragma unroll
      for (int i = 0; i < 8; ++i) u.hlf[i] = f2bf(acc[s][dt][i] * rinv[i]);
      int d = dt * 16 + l15;
      *(uint4*)(Oh[s] + (size_t)d * NPIX + q0 + hi * 8) = u.q;
    }
}

/* ---------------- host orchestration -------------------------------------- */

extern "C" void kernel_launch(void* const* d_in, const int* in_sizes, int n_in,
                              void* d_out, int out_size, void* d_ws, size_t ws_size,
                              hipStream_t stream)
{
  (void)in_sizes; (void)n_in; (void)out_size; (void)ws_size;
  const float* Xprev   = (const float*)d_in[0];
  const float* Xcurr   = (const float*)d_in[1];
  const float* flow    = (const float*)d_in[2];
  const float* net     = (const float*)d_in[3];
  const float* inp     = (const float*)d_in[4];
  const float* cc_w    = (const float*)d_in[5];
  const float* red1_w  = (const float*)d_in[6];
  const float* red1_b  = (const float*)d_in[7];
  const float* red2_w  = (const float*)d_in[8];
  const float* red2_b  = (const float*)d_in[9];
  const float* inc1_w  = (const float*)d_in[10];
  const float* inc1_b  = (const float*)d_in[11];
  const float* inc2_w  = (const float*)d_in[12];
  const float* inc2_b  = (const float*)d_in[13];
  const float* redf1_w = (const float*)d_in[14];
  const float* redf1_b = (const float*)d_in[15];
  const float* redf2_w = (const float*)d_in[16];
  const float* redf2_b = (const float*)d_in[17];
  const float* wq      = (const float*)d_in[18];
  const float* wk      = (const float*)d_in[19];
  const float* wv_flow = (const float*)d_in[20];
  const float* wv_net  = (const float*)d_in[21];
  const float* wv_inp  = (const float*)d_in[22];
  const float* p_flow  = (const float*)d_in[23];
  const float* p_net   = (const float*)d_in[24];
  const float* p_inp   = (const float*)d_in[25];
  const float* g_flow  = (const float*)d_in[26];
  const float* g_net   = (const float*)d_in[27];
  const float* g_inp   = (const float*)d_in[28];
  float* out = (float*)d_out;

  char* ws = (char*)d_ws;
  size_t off = 0;
  auto allocF = [&](size_t n) -> float* { float* p = (float*)(ws + off); off += ((n * 4 + 255) & ~(size_t)255); return p; };
  auto allocH = [&](size_t n) -> bf16*  { bf16*  p = (bf16*)(ws + off);  off += ((n * 2 + 255) & ~(size_t)255); return p; };

  float* W_r   = allocF(128 * 256);
  float* b_r   = allocF(256);
  float* W_eff = allocF(128 * 256);
  float* Wq_e  = allocF(512 * 256);
  float* bq    = allocF(512);
  float* Wk_e  = allocF(512 * 256);
  float* bk    = allocF(512);
  float* W_fl  = allocF(128 * 2);
  float* b_fl  = allocF(128);
  float* Wvf   = allocF(512 * 2);
  float* bvf   = allocF(512);
  float* T2    = allocF(2 * 128);
  float* b_of  = allocF(64);
  float* Wof   = allocF(2 * 512);

  bf16* WqH   = allocH(512 * 256);
  bf16* WkH   = allocH(512 * 256);
  bf16* wvnH  = allocH(512 * 128);
  bf16* wviH  = allocH(512 * 128);
  bf16* pnH   = allocH(128 * 512);
  bf16* piH   = allocH(128 * 512);
  bf16* WofH  = allocH(2 * 512);
  bf16* XpH   = allocH((size_t)BATCH * 256 * NPIX);
  bf16* XcH   = allocH((size_t)BATCH * 256 * NPIX);
  bf16* netH  = allocH((size_t)BATCH * 128 * NPIX);
  bf16* inpH  = allocH((size_t)BATCH * 128 * NPIX);
  bf16* QH    = allocH((size_t)BATCH * 512 * NPIX);
  bf16* KH    = allocH((size_t)BATCH * 512 * NPIX);
  bf16* VnH   = allocH((size_t)BATCH * 512 * NPIX);
  bf16* ViH   = allocH((size_t)BATCH * 512 * NPIX);
  bf16* VfH   = allocH((size_t)BATCH * 512 * NPIX);
  bf16* OfH   = allocH((size_t)BATCH * 512 * NPIX);
  bf16* OnH   = allocH((size_t)BATCH * 512 * NPIX);
  bf16* OiH   = allocH((size_t)BATCH * 512 * NPIX);

  auto mm = [&](float* C, const float* A, const float* B, int M, int K, int N, float a) {
    int tot = M * N;
    k_matmul<<<(tot + 255) / 256 > 1024 ? 1024 : (tot + 255) / 256, 256, 0, stream>>>(C, A, B, M, K, N, a);
  };
  auto mv = [&](float* y, const float* A, const float* x, const float* c, int M, int K, float a) {
    k_matvec<<<(M + 127) / 128, 128, 0, stream>>>(y, A, x, c, M, K, a);
  };
  auto cvt = [&](bf16* d, const float* s, size_t n) {
    int g = (int)((n + 255) / 256); if (g > 2048) g = 2048;
    k_f32_to_bf16<<<g, 256, 0, stream>>>(d, s, (int)n);
  };
  auto gemm = [&](const bf16* A, const bf16* B, const float* bias, const float* resid,
                  const float* gp, float* oF, bf16* oH, int M, int K, int N, int sM) {
    dim3 grid(N / GBN, (M + GBM - 1) / GBM);
    k_wmma_gemm<<<grid, 256, 0, stream>>>(A, B, bias, resid, gp, oF, oH, M, K, N, sM);
  };

  /* ---- fold the linear chains (tiny) ---- */
  mm(W_r, red2_w, red1_w, 128, 256, 256, 1.f);
  mv(b_r, red2_w, red1_b, red2_b, 128, 256, 1.f);
  mm(W_eff, W_r, cc_w, 128, 256, 256, 1.f);
  mm(Wq_e, wq, W_eff, 512, 128, 256, QKSCALE);
  mv(bq, wq, b_r, nullptr, 512, 128, QKSCALE);
  mm(Wk_e, wk, W_eff, 512, 128, 256, 1.f);
  mv(bk, wk, b_r, nullptr, 512, 128, 1.f);
  mm(W_fl, inc2_w, inc1_w, 128, 128, 2, 1.f);
  mv(b_fl, inc2_w, inc1_b, inc2_b, 128, 128, 1.f);
  mm(Wvf, wv_flow, W_fl, 512, 128, 2, 1.f);
  mv(bvf, wv_flow, b_fl, nullptr, 512, 128, 1.f);
  mm(T2, redf2_w, redf1_w, 2, 128, 128, 1.f);
  mv(b_of, redf2_w, redf1_b, redf2_b, 2, 128, 1.f);
  mm(Wof, T2, p_flow, 2, 128, 512, 1.f);

  /* ---- bf16 conversions ---- */
  cvt(WqH, Wq_e, 512 * 256);        cvt(WkH, Wk_e, 512 * 256);
  cvt(wvnH, wv_net, 512 * 128);     cvt(wviH, wv_inp, 512 * 128);
  cvt(pnH, p_net, 128 * 512);       cvt(piH, p_inp, 128 * 512);
  cvt(WofH, Wof, 2 * 512);
  cvt(XpH, Xprev, (size_t)BATCH * 256 * NPIX);
  cvt(XcH, Xcurr, (size_t)BATCH * 256 * NPIX);
  cvt(netH, net, (size_t)BATCH * 128 * NPIX);
  cvt(inpH, inp, (size_t)BATCH * 128 * NPIX);

  /* ---- V_flow (K=2, plain FMA) ---- */
  k_vflow<<<2048, 256, 0, stream>>>(VfH, flow, Wvf, bvf);

  /* ---- Q/K/V projections (WMMA) ---- */
  for (int b = 0; b < BATCH; ++b) {
    size_t x256 = (size_t)b * 256 * NPIX, x128 = (size_t)b * 128 * NPIX, x512 = (size_t)b * 512 * NPIX;
    gemm(WqH, XpH + x256, bq, nullptr, nullptr, nullptr, QH + x512, 512, 256, NPIX, 512);
    gemm(WkH, XcH + x256, bk, nullptr, nullptr, nullptr, KH + x512, 512, 256, NPIX, 512);
    gemm(wvnH, netH + x128, nullptr, nullptr, nullptr, nullptr, VnH + x512, 512, 128, NPIX, 512);
    gemm(wviH, inpH + x128, nullptr, nullptr, nullptr, nullptr, ViH + x512, 512, 128, NPIX, 512);
  }

  /* ---- fused flash attention, 3 V sources ---- */
  {
    dim3 grid(NPIX / 64, HEADS, BATCH);
    k_flash<<<grid, 128, 0, stream>>>(QH, KH, VfH, VnH, ViH, OfH, OnH, OiH);
  }

  /* ---- output projections + residual (WMMA) ---- */
  for (int b = 0; b < BATCH; ++b) {
    size_t x512 = (size_t)b * 512 * NPIX, x128 = (size_t)b * 128 * NPIX, x2 = (size_t)b * 2 * NPIX;
    gemm(WofH, OfH + x512, b_of, flow + x2, g_flow, out + x2, nullptr, 2, 512, NPIX, 2);
    gemm(pnH, OnH + x512, nullptr, net + x128, g_net,
         out + (size_t)BATCH * 2 * NPIX + x128, nullptr, 128, 512, NPIX, 128);
    gemm(piH, OiH + x512, nullptr, inp + x128, g_inp,
         out + (size_t)BATCH * 2 * NPIX + (size_t)BATCH * 128 * NPIX + x128, nullptr,
         128, 512, NPIX, 128);
  }
}
